// SQLNet_57818849739016
// MI455X (gfx1250) — matrix operand order repaired and checked by reading, not compiled, not run on hardware
//
#include <hip/hip_runtime.h>
#include <hip/hip_bf16.h>

// ---------------- problem constants ----------------
#define BB 64
#define LL 512
#define DD 768
#define QQ 150
#define HH 30
#define MASKEDV (-2147483648.0f)

typedef _Float16 f16;
typedef __attribute__((ext_vector_type(8)))  _Float16 v8h;
typedef __attribute__((ext_vector_type(16))) _Float16 v16h;
typedef __attribute__((ext_vector_type(8)))  float    v8f;

// ==================================================================
// f32 -> f16 transpose convert (weights): dst[c*rows + r] = src[r*cols + c]
// ==================================================================
__global__ void cvtT_f16_kernel(const float* __restrict__ src, f16* __restrict__ dst,
                                int rows, int cols)
{
    int i = blockIdx.x * 256 + threadIdx.x;
    if (i >= rows * cols) return;
    int r = i / cols, c = i - r * cols;
    dst[(size_t)c * rows + r] = (f16)src[(size_t)r * cols + c];
}

// gather cls rows -> f16 [B][D]
__global__ void cls16_kernel(const float* __restrict__ emb, f16* __restrict__ cls16)
{
    int i = blockIdx.x * 256 + threadIdx.x;
    if (i >= BB * DD) return;
    int b = i / DD, d = i - b * DD;
    cls16[i] = (f16)emb[(size_t)b * LL * DD + d];
}

// gather q_enc rows -> f16 [B*Q][D]
__global__ void qenc16_kernel(const float* __restrict__ emb, f16* __restrict__ q16)
{
    size_t i = (size_t)blockIdx.x * 256 + threadIdx.x;
    if (i >= (size_t)BB * QQ * DD) return;
    int d = (int)(i % DD);
    size_t row = i / DD;
    int b = (int)(row / QQ);
    int q = (int)(row - (size_t)b * QQ);
    q16[i] = (f16)emb[((size_t)b * LL + 1 + q) * DD + d];
}

// ==================================================================
// e[b][l] = dot(cls_b, emb[b][l]) / sqrt(D)
// ==================================================================
__global__ void e_kernel(const float* __restrict__ emb, float* __restrict__ e)
{
    int bl = blockIdx.x;
    int b  = bl >> 9;
    int l  = bl & (LL - 1);
    const float* cls = emb + (size_t)b * LL * DD;
    const float* row = emb + ((size_t)b * LL + l) * DD;
    float s = 0.f;
    for (int d = threadIdx.x; d < DD; d += 64) s += cls[d] * row[d];
    __shared__ float red[64];
    int t = threadIdx.x;
    red[t] = s; __syncthreads();
    for (int off = 32; off > 0; off >>= 1) {
        if (t < off) red[t] += red[t + off];
        __syncthreads();
    }
    if (t == 0) e[bl] = red[0] * 0.03608439182435161f;   // 1/sqrt(768)
}

// ==================================================================
// segment softmax attention -> h_enc (f16) [B*H][D]
// ==================================================================
__global__ void henc_kernel(const float* __restrict__ emb, const float* __restrict__ e,
                            const int* __restrict__ hlen, const int* __restrict__ seps,
                            f16* __restrict__ henc)
{
    int bh = blockIdx.x;
    int b  = bh / HH;
    int h  = bh - b * HH;
    f16* outp = henc + (size_t)bh * DD;

    bool validh = (h < hlen[b]);
    int start = seps[b * (HH + 1) + h] + 1;
    int end   = seps[b * (HH + 1) + h + 1];
    if (start < 0) start = 0;
    if (end > LL)  end = LL;
    int len = end - start;

    if (!validh || len <= 0) {
        for (int d = threadIdx.x; d < DD; d += 256) outp[d] = (f16)0.f;
        return;
    }

    __shared__ float w[LL];
    __shared__ float sinv;
    if (threadIdx.x == 0) {
        float m = -1e30f;
        for (int l = start; l < end; ++l) m = fmaxf(m, e[b * LL + l]);
        float ssum = 0.f;
        for (int l = start; l < end; ++l) {
            float t = expf(e[b * LL + l] - m);
            w[l - start] = t;
            ssum += t;
        }
        sinv = 1.f / ssum;
    }
    __syncthreads();
    float inv = sinv;
    for (int d = threadIdx.x; d < DD; d += 256) {
        float acc = 0.f;
        for (int l = 0; l < len; ++l)
            acc += w[l] * emb[((size_t)b * LL + start + l) * DD + d];
        outp[d] = (f16)(acc * inv);
    }
}

// ==================================================================
// WMMA GEMM.  C[M x N] = opt_relu(A[M x K] @ W[K x N] + bias)
// A: f16 row-major (lda halves), WT: f16 [N][K] (pre-transposed).
// M % 64 == 0, N % 64 == 0, K % 32 == 0 guaranteed -> no bounds checks.
// 128 threads (4 wave32) per block; 64x64 tile; wave = 2x2 x (16x16x32 WMMA).
// Double-buffered LDS; tiles staged with global_load_async_to_lds_b128
// (ASYNCcnt) on gfx1250, vectorized copies otherwise.
// ==================================================================
#define TM 64
#define TN 64
#define TK 32
#define LDP (TK + 8)   // LDS row stride in halves (80 B, 16B-aligned)

__device__ __forceinline__ void stage_tile(f16 (*As)[LDP], f16 (*Bs)[LDP],
                                           const f16* __restrict__ A, size_t lda,
                                           const f16* __restrict__ WT, size_t K,
                                           int tileM, int tileN, int k0, int tidx)
{
#if defined(__gfx1250__)
    #pragma unroll
    for (int ch = tidx; ch < 256; ch += 128) {
        int r = ch >> 2, p = ch & 3;            // 64 rows x 4 x 16B chunks
        const f16* ga = A + (size_t)(tileM + r) * lda + k0 + p * 8;
        unsigned la = (unsigned)(size_t)(void*)&As[r][p * 8];
        asm volatile("global_load_async_to_lds_b128 %0, %1, off"
                     :: "v"(la), "v"(ga) : "memory");
        const f16* gb = WT + (size_t)(tileN + r) * K + k0 + p * 8;
        unsigned lb = (unsigned)(size_t)(void*)&Bs[r][p * 8];
        asm volatile("global_load_async_to_lds_b128 %0, %1, off"
                     :: "v"(lb), "v"(gb) : "memory");
    }
#else
    #pragma unroll
    for (int ch = tidx; ch < 256; ch += 128) {
        int r = ch >> 2, p = ch & 3;
        *(uint4*)&As[r][p * 8] = *((const uint4*)(A + (size_t)(tileM + r) * lda + k0) + p);
        *(uint4*)&Bs[r][p * 8] = *((const uint4*)(WT + (size_t)(tileN + r) * K + k0) + p);
    }
#endif
}

__device__ __forceinline__ void wait_stage()
{
#if defined(__gfx1250__)
    asm volatile("s_wait_asynccnt 0x0" ::: "memory");
#endif
}

__global__ void gemm_wmma_kernel(const f16* __restrict__ A, size_t lda,
                                 const f16* __restrict__ WT,
                                 const float* __restrict__ bias,
                                 void* __restrict__ Cp,
                                 int M, int N, int K, int relu, int out16)
{
    __shared__ f16 As[2][TM][LDP];
    __shared__ f16 Bs[2][TN][LDP];

    int tidx = threadIdx.x;
    int wave = tidx >> 5;
    int lane = tidx & 31;
    int g    = lane >> 4;       // lane half
    int ln16 = lane & 15;
    int tileM = blockIdx.y * TM;
    int tileN = blockIdx.x * TN;
    int wm = (wave >> 1) * 32;
    int wn = (wave & 1) * 32;

    v8f acc[2][2] = {};

    stage_tile(As[0], Bs[0], A, lda, WT, (size_t)K, tileM, tileN, 0, tidx);
    wait_stage();
    __syncthreads();

    int nk = K / TK;
    for (int t = 0; t < nk; ++t) {
        int cur = t & 1;
        if (t + 1 < nk)   // prefetch next tile while WMMAs run on current one
            stage_tile(As[cur ^ 1], Bs[cur ^ 1], A, lda, WT, (size_t)K,
                       tileM, tileN, (t + 1) * TK, tidx);

        f16 (*Ac)[LDP] = As[cur];
        f16 (*Bc)[LDP] = Bs[cur];

        // B fragment: bf[idx] = W[k = g*16+idx][n]  (contiguous in Bc[n])
        v16h bfrag[2];
        #pragma unroll
        for (int j = 0; j < 2; ++j) {
            int n = wn + 16 * j + ln16;
            v8h lo = *(const v8h*)&Bc[n][g * 16];
            v8h hi = *(const v8h*)&Bc[n][g * 16 + 8];
            bfrag[j] = __builtin_shufflevector(lo, hi, 0,1,2,3,4,5,6,7,8,9,10,11,12,13,14,15);
        }
        // A fragment: af[idx] = A[m][k = g*16+idx]  (contiguous in Ac[m])
        v16h afrag[2];
        #pragma unroll
        for (int i = 0; i < 2; ++i) {
            int m = wm + 16 * i + ln16;
            v8h lo = *(const v8h*)&Ac[m][g * 16];
            v8h hi = *(const v8h*)&Ac[m][g * 16 + 8];
            afrag[i] = __builtin_shufflevector(lo, hi, 0,1,2,3,4,5,6,7,8,9,10,11,12,13,14,15);
        }
        #pragma unroll
        for (int i = 0; i < 2; ++i)
            #pragma unroll
            for (int j = 0; j < 2; ++j)
                acc[i][j] = __builtin_amdgcn_wmma_f32_16x16x32_f16(
                    false, afrag[i], false, bfrag[j],
                    (short)0, acc[i][j], false, false);

        wait_stage();
        __syncthreads();
    }

    // epilogue: c[r] -> row g*8 + r, col = lane%16
    float* C32 = (float*)Cp;
    f16*   C16 = (f16*)Cp;
    #pragma unroll
    for (int i = 0; i < 2; ++i) {
        #pragma unroll
        for (int j = 0; j < 2; ++j) {
            int col = tileN + wn + 16 * j + ln16;
            float bv = bias ? bias[col] : 0.f;
            #pragma unroll
            for (int r = 0; r < 8; ++r) {
                int row = tileM + wm + 16 * i + g * 8 + r;
                float val = acc[i][j][r] + bv;
                if (relu) val = val > 0.f ? val : 0.f;
                if (out16) C16[(size_t)row * N + col] = (f16)val;
                else       C32[(size_t)row * N + col] = val;
            }
        }
    }
}

// ==================================================================
// x16[o] = f16( LayerNorm(a2[arow] + b2[b2row]) * g + b )
// nb2==1: arow=o, b2row=b;  nb2==4: arow=o/4, b2row=b*4 + (o&3)
// ==================================================================
__device__ __forceinline__ float block_reduce_sum256(float v, float* red)
{
    int t = threadIdx.x;
    red[t] = v; __syncthreads();
    for (int off = 128; off > 0; off >>= 1) {
        if (t < off) red[t] += red[t + off];
        __syncthreads();
    }
    float r = red[0];
    __syncthreads();
    return r;
}

__global__ void ln_add_kernel(const float* __restrict__ a2, const float* __restrict__ b2v,
                              const float* __restrict__ gam, const float* __restrict__ bet,
                              f16* __restrict__ x16, int rowsOut, int nb2, int divisor)
{
    __shared__ float red[256];
    int o = blockIdx.x;
    if (o >= rowsOut) return;
    int arow  = (nb2 == 1) ? o : (o >> 2);
    int b     = arow / divisor;
    int b2row = (nb2 == 1) ? b : (b * 4 + (o & 3));
    const float* ap = a2 + (size_t)arow * DD;
    const float* bp = b2v + (size_t)b2row * DD;

    float s = 0.f;
    for (int d = threadIdx.x; d < DD; d += 256) s += ap[d] + bp[d];
    float mu = block_reduce_sum256(s, red) * (1.f / DD);
    float vs = 0.f;
    for (int d = threadIdx.x; d < DD; d += 256) {
        float t = ap[d] + bp[d] - mu;
        vs += t * t;
    }
    float var = block_reduce_sum256(vs, red) * (1.f / DD);
    float inv = rsqrtf(var + 1e-5f);
    f16* xp = x16 + (size_t)o * DD;
    for (int d = threadIdx.x; d < DD; d += 256) {
        float t = ap[d] + bp[d];
        xp[d] = (f16)(gam[d] * (t - mu) * inv + bet[d]);
    }
}

// ==================================================================
// out = h16 @ o2_w + o2_b  (dout <= 6); mode 1 -> MASKED fill per h_length
// ==================================================================
__global__ void out_small_kernel(const f16* __restrict__ hbuf, const float* __restrict__ o2w,
                                 const float* __restrict__ o2b, float* __restrict__ outp,
                                 int M, int dout, int mode, const int* __restrict__ hlen)
{
    __shared__ float red[8 * 256];
    int row = blockIdx.x;
    if (row >= M) return;
    const f16* hr = hbuf + (size_t)row * DD;
    float acc[8];
    for (int c = 0; c < dout; ++c) acc[c] = 0.f;
    for (int k = threadIdx.x; k < DD; k += 256) {
        float hv = (float)hr[k];
        for (int c = 0; c < dout; ++c) acc[c] += hv * o2w[(size_t)k * dout + c];
    }
    for (int c = 0; c < dout; ++c) red[c * 256 + threadIdx.x] = acc[c];
    __syncthreads();
    for (int off = 128; off > 0; off >>= 1) {
        if (threadIdx.x < off)
            for (int c = 0; c < dout; ++c)
                red[c * 256 + threadIdx.x] += red[c * 256 + threadIdx.x + off];
        __syncthreads();
    }
    if (threadIdx.x < dout) {
        float val = red[threadIdx.x * 256] + o2b[threadIdx.x];
        if (mode == 1) {
            int b = row / HH;
            int h = row - b * HH;
            if (h >= hlen[b]) val = MASKEDV;
        }
        outp[(size_t)row * dout + threadIdx.x] = val;
    }
}

// ==================================================================
// flags -> spans -> v_enc (f16 segment means of q_enc)
// ==================================================================
__global__ void venc_kernel(const float* __restrict__ emb, const float* __restrict__ wval,
                            const int* __restrict__ qlen, f16* __restrict__ venc)
{
    int b = blockIdx.x;
    __shared__ int   seg[QQ];
    __shared__ float cnt[4];
    if (threadIdx.x == 0) {
        int ql = qlen[b];
        int prev = 0, span = -1;
        float c[4] = {0.f, 0.f, 0.f, 0.f};
        for (int q = 0; q < QQ; ++q) {
            const float* wv = wval + ((size_t)b * QQ + q) * 2;
            int f = (wv[1] > wv[0]) ? 1 : 0;   // argmax, tie -> 0
            if (q >= ql) f = 0;
            int s = -1;
            if (f == 1) {
                if (prev == 0) span++;
                if (span >= 0 && span < 4) s = span;
            }
            seg[q] = s;
            prev = f;
            if (s >= 0) c[s] += 1.f;
        }
        for (int j = 0; j < 4; ++j) cnt[j] = fmaxf(c[j], 1.f);
    }
    __syncthreads();
    for (int j = 0; j < 4; ++j) {
        float inv = 1.f / cnt[j];
        for (int d = threadIdx.x; d < DD; d += 256) {
            float s = 0.f;
            for (int q = 0; q < QQ; ++q)
                if (seg[q] == j) s += emb[((size_t)b * LL + 1 + q) * DD + d];
            venc[((size_t)b * 4 + j) * DD + d] = (f16)(s * inv);
        }
    }
}

// ==================================================================
// host side
// ==================================================================
extern "C" void kernel_launch(void* const* d_in, const int* in_sizes, int n_in,
                              void* d_out, int out_size, void* d_ws, size_t ws_size,
                              hipStream_t stream)
{
    const float* emb  = (const float*)d_in[0];
    const int*   qlen = (const int*)d_in[1];
    const int*   hlen = (const int*)d_in[2];
    const int*   seps = (const int*)d_in[3];
    auto P = [&](int head, int j) -> const float* {
        return (const float*)d_in[4 + head * 10 + j];
    };

    // ---------------- workspace layout (bytes) ----------------
    char* wsb = (char*)d_ws;
    size_t off = 0;
    auto alloc = [&](size_t bytes) { void* p = wsb + off; off += (bytes + 255) & ~(size_t)255; return p; };
    float* e      = (float*)alloc((size_t)BB * LL * 4);            // scores
    f16*   cls16  = (f16*)  alloc((size_t)BB * DD * 2);
    f16*   henc16 = (f16*)  alloc((size_t)BB * HH * DD * 2);
    f16*   qenc16 = (f16*)  alloc((size_t)BB * QQ * DD * 2);
    f16*   venc16 = (f16*)  alloc((size_t)BB * 4 * DD * 2);
    float* b2     = (float*)alloc((size_t)BB * DD * 4);
    float* b2m    = (float*)alloc((size_t)BB * 4 * DD * 4);
    float* bufA   = (float*)alloc((size_t)BB * QQ * DD * 4);       // a2 (f32, max 9600 rows)
    f16*   x16    = (f16*)  alloc((size_t)BB * QQ * DD * 2);       // LN output
    f16*   hid16  = (f16*)  alloc((size_t)BB * QQ * DD * 2);       // relu hidden
    f16*   w16    = (f16*)  alloc((size_t)27 * DD * DD * 2);       // transposed f16 weights
    float* outp   = (float*)d_out;

    auto WT = [&](int head, int slot) -> f16* {                    // slot 0:pa 1:pb 2:o1
        return w16 + ((size_t)head * 3 + slot) * DD * DD;
    };

    const long OFF_SNUM = 0, OFF_SCOL = 256, OFF_SAGG = 2176, OFF_WNUM = 13696,
               OFF_WCONN = 14016, OFF_WCOL = 14144, OFF_WOP = 16064,
               OFF_WVAL = 23744, OFF_WMATCH = 42944;

    auto GEMM = [&](const f16* A, size_t lda, const f16* WTm, const float* bias,
                    void* C, int M, int N, int K, int relu, int out16) {
        dim3 grid(N / TN, M / TM);
        gemm_wmma_kernel<<<grid, 128, 0, stream>>>(A, lda, WTm, bias, C, M, N, K, relu, out16);
    };

    // -------- stage 0: weight transpose-convert + activation gathers --------
    {
        int nblk = (DD * DD + 255) / 256;
        for (int h = 0; h < 9; ++h) {
            cvtT_f16_kernel<<<nblk, 256, 0, stream>>>(P(h, 0), WT(h, 0), DD, DD); // pa_w
            cvtT_f16_kernel<<<nblk, 256, 0, stream>>>(P(h, 2), WT(h, 1), DD, DD); // pb_w
            cvtT_f16_kernel<<<nblk, 256, 0, stream>>>(P(h, 6), WT(h, 2), DD, DD); // o1_w
        }
        cls16_kernel<<<(BB * DD + 255) / 256, 256, 0, stream>>>(emb, cls16);
        qenc16_kernel<<<(BB * QQ * DD + 255) / 256, 256, 0, stream>>>(emb, qenc16);
    }

    // -------- attention path --------
    e_kernel<<<BB * LL, 64, 0, stream>>>(emb, e);
    henc_kernel<<<BB * HH, 256, 0, stream>>>(emb, e, hlen, seps, henc16);

    // -------- cls-only heads: select_num(0,d4) where_num(4,d5) where_conn_op(3,d2) --------
    {
        int  heads[3] = {0, 4, 3};
        int  douts[3] = {4, 5, 2};
        long offs[3]  = {OFF_SNUM, OFF_WNUM, OFF_WCONN};
        for (int i = 0; i < 3; ++i) {
            int hd = heads[i];
            GEMM(cls16, DD, WT(hd, 2), P(hd, 7), hid16, BB, DD, DD, 1, 1);
            out_small_kernel<<<BB, 256, 0, stream>>>(hid16, P(hd, 8), P(hd, 9),
                                                     outp + offs[i], BB, douts[i], 0, hlen);
        }
    }

    // -------- pair heads on h_enc with cls as b --------
    {
        int  heads[4] = {1, 2, 5, 6};
        int  douts[4] = {1, 6, 1, 4};
        long offs[4]  = {OFF_SCOL, OFF_SAGG, OFF_WCOL, OFF_WOP};
        int  modes[4] = {1, 0, 1, 0};
        for (int i = 0; i < 4; ++i) {
            int hd = heads[i];
            GEMM(cls16, DD, WT(hd, 1), P(hd, 3), b2, BB, DD, DD, 0, 0);           // b2 = cls@pb
            GEMM(henc16, DD, WT(hd, 0), P(hd, 1), bufA, BB * HH, DD, DD, 0, 0);   // a2 = h_enc@pa
            ln_add_kernel<<<BB * HH, 256, 0, stream>>>(bufA, b2, P(hd, 4), P(hd, 5),
                                                       x16, BB * HH, 1, HH);
            GEMM(x16, DD, WT(hd, 2), P(hd, 7), hid16, BB * HH, DD, DD, 1, 1);     // relu(x@o1)
            out_small_kernel<<<BB * HH, 256, 0, stream>>>(hid16, P(hd, 8), P(hd, 9),
                                                          outp + offs[i], BB * HH,
                                                          douts[i], modes[i], hlen);
        }
    }

    // -------- where_val (7): pair(q_enc, cls) -> spans -> v_enc --------
    {
        int hd = 7;
        GEMM(cls16, DD, WT(hd, 1), P(hd, 3), b2, BB, DD, DD, 0, 0);
        GEMM(qenc16, DD, WT(hd, 0), P(hd, 1), bufA, BB * QQ, DD, DD, 0, 0);
        ln_add_kernel<<<BB * QQ, 256, 0, stream>>>(bufA, b2, P(hd, 4), P(hd, 5),
                                                   x16, BB * QQ, 1, QQ);
        GEMM(x16, DD, WT(hd, 2), P(hd, 7), hid16, BB * QQ, DD, DD, 1, 1);
        out_small_kernel<<<BB * QQ, 256, 0, stream>>>(hid16, P(hd, 8), P(hd, 9),
                                                      outp + OFF_WVAL, BB * QQ, 2, 0, hlen);
        venc_kernel<<<BB, 256, 0, stream>>>(emb, outp + OFF_WVAL, qlen, venc16);
    }

    // -------- where_match (8): pair(h_enc, v_enc), x expands 4x --------
    {
        int hd = 8;
        GEMM(venc16, DD, WT(hd, 1), P(hd, 3), b2m, BB * 4, DD, DD, 0, 0);          // b2 = v_enc@pb
        GEMM(henc16, DD, WT(hd, 0), P(hd, 1), bufA, BB * HH, DD, DD, 0, 0);        // a2 = h_enc@pa
        ln_add_kernel<<<BB * HH * 4, 256, 0, stream>>>(bufA, b2m, P(hd, 4), P(hd, 5),
                                                       x16, BB * HH * 4, 4, HH);
        GEMM(x16, DD, WT(hd, 2), P(hd, 7), hid16, BB * HH * 4, DD, DD, 1, 1);
        out_small_kernel<<<BB * HH * 4, 256, 0, stream>>>(hid16, P(hd, 8), P(hd, 9),
                                                          outp + OFF_WMATCH, BB * HH * 4,
                                                          1, 0, hlen);
    }
}
